// AttentionPairBias_23124103922397
// MI455X (gfx1250) — matrix operand order repaired and checked by reading, not compiled, not run on hardware
//
#include <hip/hip_runtime.h>

#define NTOK 1024
#define DIMF 1024
#define NHEAD 16
#define DHEAD 64
#define DPAIR 128
#define NN (NTOK * NTOK)

typedef __attribute__((ext_vector_type(16))) __bf16 v16bf;
typedef __attribute__((ext_vector_type(8)))  __bf16 v8bf;
typedef __attribute__((ext_vector_type(4)))  __bf16 v4bf;
typedef __attribute__((ext_vector_type(8)))  float  v8f;

__device__ __forceinline__ v8f zero8() {
  v8f z = {0.f, 0.f, 0.f, 0.f, 0.f, 0.f, 0.f, 0.f};
  return z;
}

__device__ __forceinline__ v8f wmma_bf16(v16bf a, v16bf b, v8f c) {
  // D = A(16x32 bf16) * B(32x16 bf16) + C(16x16 f32)
  return __builtin_amdgcn_wmma_f32_16x16x32_bf16(false, a, false, b, (short)0, c,
                                                 false, false);
}

// A-fragment (16x32 bf16) from a row-major tile (K contiguous, element stride 1).
// ISA: lanes 0-15 row M=lane, K in {0..7,16..23}; lanes 16-31 same M set, K+8.
// Per lane: two contiguous 16B chunks -> ds_load_b128 / global_load_b128.
__device__ __forceinline__ v16bf frag_a_rows(const __bf16* p, int mstride,
                                             int lane) {
  int m = lane & 15;
  int kofs = (lane >> 4) * 8;
  const __bf16* q = p + m * mstride + kofs;
  v8bf lo = *(const v8bf*)(q);       // K = kofs .. kofs+7
  v8bf hi = *(const v8bf*)(q + 16);  // K = 16+kofs .. 23+kofs
  v16bf a;
#pragma unroll
  for (int e = 0; e < 8; ++e) {
    a[e] = lo[e];
    a[8 + e] = hi[e];
  }
  return a;
}

// B-fragment (32x16 bf16) from a COLUMN-major tile: p[n*nstride + k].
// Lanes 0-15: N=lane, K in {0..7,16..23}; lanes 16-31: N=lane-16, K+8.
// Per lane: two contiguous 16B chunks -> two 16B LDS loads.
__device__ __forceinline__ v16bf frag_b_cols(const __bf16* p, int nstride,
                                             int lane) {
  int n = lane & 15;
  int kb = (lane >> 4) * 8;
  const __bf16* q = p + n * nstride + kb;
  v8bf lo = *(const v8bf*)(q);
  v8bf hi = *(const v8bf*)(q + 16);
  v16bf b;
#pragma unroll
  for (int e = 0; e < 8; ++e) {
    b[e] = lo[e];
    b[8 + e] = hi[e];
  }
  return b;
}

// ---------------------------------------------------------------------------
// Kernel 1: LayerNorm(pairwise) @ W_bias -> bias[h][i][j] (fp32 in ws).
// One wave = 16 pairs: shuffle-reduce mean/var, write normalized bf16 tile to
// LDS, then 4 chained WMMAs against loop-invariant W_bias fragments.
// ---------------------------------------------------------------------------
__global__ __launch_bounds__(256) void bias_kernel(
    const float* __restrict__ pair, const float* __restrict__ gamma,
    const float* __restrict__ beta, const float* __restrict__ Wb,
    float* __restrict__ bias) {
  __shared__ __bf16 sWbT[NHEAD * 136];   // col-major: [head n][feature k]
  __shared__ float sG[DPAIR];
  __shared__ float sBt[DPAIR];
  __shared__ __bf16 sA[8][16 * 136];     // per-wave normalized 16x128 tile
  __shared__ float sTile[8][NHEAD * 16]; // per-wave [h][j] staging for stores
  int tid = threadIdx.x;
#pragma unroll
  for (int l = 0; l < 8; ++l) {
    int idx = tid + l * 256;  // 2048 = 128 k * 16 n
    int n = idx & 15, k = idx >> 4;
    sWbT[n * 136 + k] = (__bf16)Wb[k * NHEAD + n];
  }
  if (tid < DPAIR) {
    sG[tid] = gamma[tid];
    sBt[tid] = beta[tid];
  }
  __syncthreads();
  int lane = tid & 31, wv = tid >> 5;
  // W_bias B-fragments: invariant across the whole kernel, keep in registers
  v16bf bW0 = frag_b_cols(sWbT + 0, 136, lane);
  v16bf bW1 = frag_b_cols(sWbT + 32, 136, lane);
  v16bf bW2 = frag_b_cols(sWbT + 64, 136, lane);
  v16bf bW3 = frag_b_cols(sWbT + 96, 136, lane);
  __bf16* sAw = sA[wv];
  int c0 = lane * 4;
  int n = lane & 15, mofs = (lane >> 4) * 8;
  int gw = blockIdx.x * 8 + wv;
  int nW = gridDim.x * 8;
  for (int t = gw; t < NTOK * 64; t += nW) {
    int i = t >> 6;
    int j0 = (t & 63) << 4;
#pragma unroll 1
    for (int jj = 0; jj < 16; ++jj) {
      const float4 xv =
          *(const float4*)(pair + ((size_t)i * NTOK + j0 + jj) * DPAIR + c0);
      float s = xv.x + xv.y + xv.z + xv.w;
      float ss = xv.x * xv.x + xv.y * xv.y + xv.z * xv.z + xv.w * xv.w;
#pragma unroll
      for (int m = 1; m < 32; m <<= 1) {
        s += __shfl_xor(s, m);
        ss += __shfl_xor(ss, m);
      }
      float mean = s * (1.0f / DPAIR);
      float var = ss * (1.0f / DPAIR) - mean * mean;
      float rs = rsqrtf(var + 1e-5f);
      v4bf tn;
      tn[0] = (__bf16)((xv.x - mean) * rs * sG[c0 + 0] + sBt[c0 + 0]);
      tn[1] = (__bf16)((xv.y - mean) * rs * sG[c0 + 1] + sBt[c0 + 1]);
      tn[2] = (__bf16)((xv.z - mean) * rs * sG[c0 + 2] + sBt[c0 + 2]);
      tn[3] = (__bf16)((xv.w - mean) * rs * sG[c0 + 3] + sBt[c0 + 3]);
      *(v4bf*)(sAw + jj * 136 + c0) = tn;  // 8B aligned store
    }
    // C(16 pairs x 16 heads) = pn(16x128) @ W_bias(128x16)
    v8f c = zero8();
    c = wmma_bf16(frag_a_rows(sAw + 0, 136, lane), bW0, c);
    c = wmma_bf16(frag_a_rows(sAw + 32, 136, lane), bW1, c);
    c = wmma_bf16(frag_a_rows(sAw + 64, 136, lane), bW2, c);
    c = wmma_bf16(frag_a_rows(sAw + 96, 136, lane), bW3, c);
    // C layout: row = pair (mofs+r), col = head n -> transit LDS for
    // coalesced per-head global stores (same-wave LDS ops are in order)
#pragma unroll
    for (int r = 0; r < 8; ++r) sTile[wv][n * 16 + mofs + r] = c[r];
#pragma unroll
    for (int hh = 0; hh < 8; ++hh) {
      int h = hh * 2 + (lane >> 4);
      int jc = lane & 15;
      bias[(size_t)h * NN + (size_t)i * NTOK + j0 + jc] = sTile[wv][h * 16 + jc];
    }
  }
}

// ---------------------------------------------------------------------------
// Kernel 2: fused projections X @ {Wq,Wk,Wv,Wg}. Block tile 128x128, BK=32.
// A row-major bf16 (stride 40), B column-major bf16 (stride 40). Each wave:
// 16x128 output = 8 WMMA accumulators.
// ---------------------------------------------------------------------------
__global__ __launch_bounds__(256) void proj_kernel(
    const float* __restrict__ X, const float* __restrict__ Wq,
    const float* __restrict__ Wk, const float* __restrict__ Wv,
    const float* __restrict__ Wg, const float* __restrict__ bg,
    __bf16* __restrict__ Qb, __bf16* __restrict__ Kb, __bf16* __restrict__ Vb,
    float* __restrict__ Gf) {
  __shared__ __bf16 sA[128 * 40];
  __shared__ __bf16 sBt[128 * 40];  // column-major: [n][k]
  int tid = threadIdx.x, lane = tid & 31, wv = tid >> 5;
  int m0 = blockIdx.x * 128;
  int nb = blockIdx.y;
  int which = nb >> 3;
  int ncol0 = (nb & 7) * 128;
  const float* W = (which == 0) ? Wq : (which == 1) ? Wk : (which == 2) ? Wv : Wg;
  v8f acc[8];
#pragma unroll
  for (int q = 0; q < 8; ++q) acc[q] = zero8();
  for (int kb = 0; kb < DIMF; kb += 32) {
#pragma unroll
    for (int l2 = 0; l2 < 4; ++l2) {  // stage 128x32 of X (row-major)
      int idx = tid + l2 * 256;
      int r = idx >> 3, c4 = (idx & 7) * 4;
      float4 t4 = *(const float4*)(X + (size_t)(m0 + r) * DIMF + kb + c4);
      v4bf tb;
      tb[0] = (__bf16)t4.x; tb[1] = (__bf16)t4.y;
      tb[2] = (__bf16)t4.z; tb[3] = (__bf16)t4.w;
      *(v4bf*)(sA + r * 40 + c4) = tb;
    }
#pragma unroll
    for (int l2 = 0; l2 < 4; ++l2) {  // stage 32x128 of W (column-major)
      int idx = tid + l2 * 256;
      int nn = idx & 127;
      int r = (idx >> 7) * 4;
      const float* wp = W + (size_t)(kb + r) * DIMF + ncol0 + nn;
      v4bf tb;
      tb[0] = (__bf16)wp[0];
      tb[1] = (__bf16)wp[DIMF];
      tb[2] = (__bf16)wp[2 * DIMF];
      tb[3] = (__bf16)wp[3 * DIMF];
      *(v4bf*)(sBt + nn * 40 + r) = tb;
    }
    __syncthreads();
    v16bf a = frag_a_rows(sA + wv * 16 * 40, 40, lane);
#pragma unroll
    for (int q = 0; q < 8; ++q) {
      v16bf b = frag_b_cols(sBt + q * 16 * 40, 40, lane);
      acc[q] = wmma_bf16(a, b, acc[q]);
    }
    __syncthreads();
  }
  int mrow0 = m0 + wv * 16 + (lane >> 4) * 8;
  int n = lane & 15;
#pragma unroll
  for (int q = 0; q < 8; ++q) {
#pragma unroll
    for (int r = 0; r < 8; ++r) {
      int row = mrow0 + r;
      int col = ncol0 + q * 16 + n;
      float val = acc[q][r];
      if (which == 0)
        Qb[(size_t)row * DIMF + col] = (__bf16)(val * 0.125f);  // fold 1/sqrt(64)
      else if (which == 1)
        Kb[(size_t)row * DIMF + col] = (__bf16)val;
      else if (which == 2)
        Vb[(size_t)row * DIMF + col] = (__bf16)val;
      else {
        float z = val + bg[col];
        Gf[(size_t)row * DIMF + col] = 1.0f / (1.0f + __expf(-z));
      }
    }
  }
}

// ---------------------------------------------------------------------------
// Kernel 3: flash attention with additive pair bias. One wave = 16 query rows
// of one head; block stages 32-token K (row-major) and V (transposed) chunks.
// ---------------------------------------------------------------------------
__global__ __launch_bounds__(256) void attn_kernel(
    const __bf16* __restrict__ Qb, const __bf16* __restrict__ Kb,
    const __bf16* __restrict__ Vb, const float* __restrict__ bias,
    float* __restrict__ Ob) {
  __shared__ __bf16 sK[32 * 72];      // [token][d] row-major (frag-contiguous)
  __shared__ __bf16 sVt[64 * 40];     // [d][token] transposed (frag-contiguous)
  __shared__ __bf16 sP[8][16 * 40];   // per-wave P transit tile
  int tid = threadIdx.x, lane = tid & 31, wv = tid >> 5;
  int h = blockIdx.y;
  int i0 = blockIdx.x * 128 + wv * 16;
  int n = lane & 15;
  int mofs = (lane >> 4) * 8;

  const __bf16* qbase = Qb + (size_t)i0 * DIMF + h * DHEAD;
  v16bf aQ0 = frag_a_rows(qbase, DIMF, lane);       // d = 0..31
  v16bf aQ1 = frag_a_rows(qbase + 32, DIMF, lane);  // d = 32..63

  const float* bp = bias + (size_t)h * NN + (size_t)(i0 + mofs) * NTOK;

  float m_r[8], l_r[8];
  v8f oA[4];
#pragma unroll
  for (int r = 0; r < 8; ++r) {
    m_r[r] = -1e30f;
    l_r[r] = 0.0f;
  }
#pragma unroll
  for (int q = 0; q < 4; ++q) oA[q] = zero8();

  int cr = tid >> 3;       // K staging: row 0..31
  int cc = (tid & 7) * 8;  // K staging: col 0..56
  for (int jb = 0; jb < NTOK; jb += 32) {
    // K: 32 tokens x 64 d, bit-copy 16B vectors, row-major
    *(float4*)(sK + cr * 72 + cc) =
        *(const float4*)(Kb + (size_t)(jb + cr) * DIMF + h * DHEAD + cc);
    // V: transposed into [d][token], 4 tokens per thread -> 8B b64 stores
#pragma unroll
    for (int l2 = 0; l2 < 2; ++l2) {
      int idx = tid + l2 * 256;  // 512 = 64 d * 8 token-groups
      int d = idx & 63;
      int tk = (idx >> 6) * 4;
      const __bf16* vp = Vb + (size_t)(jb + tk) * DIMF + h * DHEAD + d;
      v4bf tb;
      tb[0] = vp[0];
      tb[1] = vp[DIMF];
      tb[2] = vp[2 * DIMF];
      tb[3] = vp[3 * DIMF];
      *(v4bf*)(sVt + d * 40 + tk) = tb;
    }
    __syncthreads();

    float st0[8], st1[8];
    {  // S tile, j = jb..jb+15 : Q(16x64) * K^T(64x16)
      v8f s = zero8();
      s = wmma_bf16(aQ0, frag_b_cols(sK + 0, 72, lane), s);
      s = wmma_bf16(aQ1, frag_b_cols(sK + 32, 72, lane), s);
#pragma unroll
      for (int r = 0; r < 8; ++r) st0[r] = s[r] + bp[(size_t)r * NTOK + jb + n];
    }
    {  // S tile, j = jb+16..jb+31
      v8f s = zero8();
      s = wmma_bf16(aQ0, frag_b_cols(sK + 16 * 72 + 0, 72, lane), s);
      s = wmma_bf16(aQ1, frag_b_cols(sK + 16 * 72 + 32, 72, lane), s);
#pragma unroll
      for (int r = 0; r < 8; ++r)
        st1[r] = s[r] + bp[(size_t)r * NTOK + jb + 16 + n];
    }
    // online softmax over this 32-column chunk (16-lane halves == C row split)
#pragma unroll
    for (int r = 0; r < 8; ++r) {
      float mx = fmaxf(st0[r], st1[r]);
      mx = fmaxf(mx, __shfl_xor(mx, 1));
      mx = fmaxf(mx, __shfl_xor(mx, 2));
      mx = fmaxf(mx, __shfl_xor(mx, 4));
      mx = fmaxf(mx, __shfl_xor(mx, 8));
      float mn = fmaxf(m_r[r], mx);
      float f = __expf(m_r[r] - mn);
      float p0 = __expf(st0[r] - mn);
      float p1 = __expf(st1[r] - mn);
      float sm = p0 + p1;
      sm += __shfl_xor(sm, 1);
      sm += __shfl_xor(sm, 2);
      sm += __shfl_xor(sm, 4);
      sm += __shfl_xor(sm, 8);
      l_r[r] = l_r[r] * f + sm;
      m_r[r] = mn;
      st0[r] = p0;
      st1[r] = p1;
#pragma unroll
      for (int q = 0; q < 4; ++q) oA[q][r] *= f;
    }
    // transit P (16x32) through per-wave LDS tile into A-fragment layout
    __bf16* pp = sP[wv];
#pragma unroll
    for (int r = 0; r < 8; ++r) {
      pp[(mofs + r) * 40 + n] = (__bf16)st0[r];
      pp[(mofs + r) * 40 + 16 + n] = (__bf16)st1[r];
    }
    v16bf aP = frag_a_rows(pp, 40, lane);
#pragma unroll
    for (int q = 0; q < 4; ++q) {
      v16bf bv = frag_b_cols(sVt + q * 16 * 40, 40, lane);
      oA[q] = wmma_bf16(aP, bv, oA[q]);  // O += P(16x32) * V(32x64)
    }
    __syncthreads();
  }
#pragma unroll
  for (int r = 0; r < 8; ++r) {
    float inv = 1.0f / l_r[r];
    int row = i0 + mofs + r;
#pragma unroll
    for (int q = 0; q < 4; ++q) {
      Ob[(size_t)row * DIMF + h * DHEAD + q * 16 + n] = oA[q][r] * inv;
    }
  }
}

// ---------------------------------------------------------------------------
// Kernel 4: out = (O .* gates) @ Wo. Block tile 128x128, fp32 out.
// ---------------------------------------------------------------------------
__global__ __launch_bounds__(256) void outproj_kernel(
    const float* __restrict__ O, const float* __restrict__ G,
    const float* __restrict__ Wo, float* __restrict__ out) {
  __shared__ __bf16 sA[128 * 40];
  __shared__ __bf16 sBt[128 * 40];
  int tid = threadIdx.x, lane = tid & 31, wv = tid >> 5;
  int m0 = blockIdx.x * 128;
  int ncol0 = blockIdx.y * 128;
  v8f acc[8];
#pragma unroll
  for (int q = 0; q < 8; ++q) acc[q] = zero8();
  for (int kb = 0; kb < DIMF; kb += 32) {
#pragma unroll
    for (int l2 = 0; l2 < 4; ++l2) {  // stage gated O (row-major)
      int idx = tid + l2 * 256;
      int r = idx >> 3, c4 = (idx & 7) * 4;
      size_t off = (size_t)(m0 + r) * DIMF + kb + c4;
      float4 t4 = *(const float4*)(O + off);
      float4 g4 = *(const float4*)(G + off);
      v4bf tb;
      tb[0] = (__bf16)(t4.x * g4.x);
      tb[1] = (__bf16)(t4.y * g4.y);
      tb[2] = (__bf16)(t4.z * g4.z);
      tb[3] = (__bf16)(t4.w * g4.w);
      *(v4bf*)(sA + r * 40 + c4) = tb;
    }
#pragma unroll
    for (int l2 = 0; l2 < 4; ++l2) {  // stage 32x128 of Wo (column-major)
      int idx = tid + l2 * 256;
      int nn = idx & 127;
      int r = (idx >> 7) * 4;
      const float* wp = Wo + (size_t)(kb + r) * DIMF + ncol0 + nn;
      v4bf tb;
      tb[0] = (__bf16)wp[0];
      tb[1] = (__bf16)wp[DIMF];
      tb[2] = (__bf16)wp[2 * DIMF];
      tb[3] = (__bf16)wp[3 * DIMF];
      *(v4bf*)(sBt + nn * 40 + r) = tb;
    }
    __syncthreads();
    v16bf a = frag_a_rows(sA + wv * 16 * 40, 40, lane);
#pragma unroll
    for (int q = 0; q < 8; ++q) {
      v16bf b = frag_b_cols(sBt + q * 16 * 40, 40, lane);
      acc[q] = wmma_bf16(a, b, acc[q]);
    }
    __syncthreads();
  }
  int mrow0 = m0 + wv * 16 + (lane >> 4) * 8;
  int n = lane & 15;
#pragma unroll
  for (int q = 0; q < 8; ++q) {
#pragma unroll
    for (int r = 0; r < 8; ++r) {
      out[(size_t)(mrow0 + r) * DIMF + ncol0 + q * 16 + n] = acc[q][r];
    }
  }
}

extern "C" void kernel_launch(void* const* d_in, const int* in_sizes, int n_in,
                              void* d_out, int out_size, void* d_ws,
                              size_t ws_size, hipStream_t stream) {
  (void)in_sizes; (void)n_in; (void)out_size; (void)ws_size;
  const float* x   = (const float*)d_in[0];   // single_repr  [1024,1024]
  const float* pr  = (const float*)d_in[1];   // pairwise     [1024,1024,128]
  const float* lng = (const float*)d_in[2];   // ln_gamma [128]
  const float* lnb = (const float*)d_in[3];   // ln_beta  [128]
  const float* wb  = (const float*)d_in[4];   // W_bias [128,16]
  const float* wq  = (const float*)d_in[5];
  const float* wk  = (const float*)d_in[6];
  const float* wvp = (const float*)d_in[7];
  const float* wg  = (const float*)d_in[8];
  const float* bg  = (const float*)d_in[9];
  const float* wo  = (const float*)d_in[10];
  float* out = (float*)d_out;

  char* w = (char*)d_ws;
  float*  bias = (float*)(w);                 // 64 MB fp32 [16][1024][1024]
  __bf16* Qb   = (__bf16*)(w + 67108864);     //  2 MB bf16 (pre-scaled by 1/8)
  __bf16* Kb   = (__bf16*)(w + 69206016);     //  2 MB
  __bf16* Vb   = (__bf16*)(w + 71303168);     //  2 MB
  float*  Gf   = (float*)(w + 73400320);      //  4 MB sigmoid gates
  float*  Ob   = (float*)(w + 77594624);      //  4 MB attention output

  bias_kernel<<<dim3(1024), dim3(256), 0, stream>>>(pr, lng, lnb, wb, bias);
  proj_kernel<<<dim3(8, 32), dim3(256), 0, stream>>>(x, wq, wk, wvp, wg, bg, Qb,
                                                     Kb, Vb, Gf);
  attn_kernel<<<dim3(8, 16), dim3(256), 0, stream>>>(Qb, Kb, Vb, bias, Ob);
  outproj_kernel<<<dim3(8, 8), dim3(256), 0, stream>>>(Ob, Gf, wo, out);
}
